// Self_Attn_50354196579077
// MI455X (gfx1250) — matrix operand order repaired and checked by reading, not compiled, not run on hardware
//
#include <hip/hip_runtime.h>

typedef __attribute__((ext_vector_type(16))) _Float16 v16h;
typedef __attribute__((ext_vector_type(8)))  _Float16 v8h;
typedef __attribute__((ext_vector_type(8)))  float    v8f;
typedef __attribute__((ext_vector_type(4)))  int      v4i;

#define BN   4
#define CC   256
#define CI   32
#define NPIX 4096

#if defined(__has_builtin)
#if __has_builtin(__builtin_amdgcn_global_load_async_to_lds_b128) && \
    __has_builtin(__builtin_amdgcn_s_wait_asynccnt)
#define USE_ASYNC_LDS 1
#endif
#endif

// lane <-> lane^16 full-width swap (wave32): group-of-32 ds_swizzle, xor=0x10, and=0x1f
__device__ __forceinline__ float swap16f(float v) {
  return __int_as_float(__builtin_amdgcn_ds_swizzle(__float_as_int(v), 0x401F));
}

__device__ __forceinline__ float fast_exp2(float x) {
#if defined(__has_builtin)
#if __has_builtin(__builtin_amdgcn_exp2f)
  return __builtin_amdgcn_exp2f(x);
#else
  return exp2f(x);
#endif
#else
  return exp2f(x);
#endif
}

__device__ __forceinline__ v16h cat16(v8h a, v8h b) {
  v16h r;
#pragma unroll
  for (int i = 0; i < 8; ++i) { r[i] = a[i]; r[i + 8] = b[i]; }
  return r;
}

// one 16-byte global -> LDS copy (async on gfx1250 toolchains that expose it)
__device__ __forceinline__ void copy16(const _Float16* g, _Float16* l) {
#ifdef USE_ASYNC_LDS
  __builtin_amdgcn_global_load_async_to_lds_b128(
      (__attribute__((address_space(1))) v4i*)g,
      (__attribute__((address_space(3))) v4i*)l, 0, 0);
#else
  *(v8h*)l = *(const v8h*)g;
#endif
}

__device__ __forceinline__ void copy_fence() {
#ifdef USE_ASYNC_LDS
  __builtin_amdgcn_s_wait_asynccnt(0);
#endif
  __syncthreads();
}

// ---------------- Phase 1: 1x1-conv projections f,g,h (fp32 -> f16 operands) ---------
// thread t <-> pixel n; loop c outer (coalesced x reads, scalar W broadcast), 32 accums.
// f,g stored transposed [b][n][32]; h stored row-major [b][32][m].
// fT is pre-scaled by log2(e) so the attention softmax can use raw v_exp_f32 (base-2).
__global__ __launch_bounds__(256)
void proj_kernel(const float* __restrict__ x,
                 const float* __restrict__ Wf, const float* __restrict__ bf,
                 const float* __restrict__ Wg, const float* __restrict__ bg,
                 const float* __restrict__ Wh, const float* __restrict__ bh,
                 _Float16* __restrict__ fT, _Float16* __restrict__ gT,
                 _Float16* __restrict__ hRM) {
  const int n   = blockIdx.x * 256 + threadIdx.x;
  const int mat = blockIdx.y;
  const int b   = blockIdx.z;
  const float* W    = (mat == 0) ? Wf : (mat == 1) ? Wg : Wh;
  const float* bias = (mat == 0) ? bf : (mat == 1) ? bg : bh;
  const float* xb = x + (size_t)b * CC * NPIX;
  float acc[CI];
#pragma unroll
  for (int o = 0; o < CI; ++o) acc[o] = bias[o];
  for (int c = 0; c < CC; ++c) {
    float xv = xb[(size_t)c * NPIX + n];
#pragma unroll
    for (int o = 0; o < CI; ++o) acc[o] = fmaf(W[o * CC + c], xv, acc[o]);
  }
  if (mat < 2) {
    const float scl = (mat == 0) ? 1.44269504088896f : 1.0f;  // fold log2e into f
    _Float16* dst = ((mat == 0) ? fT : gT) + ((size_t)b * NPIX + n) * CI;
#pragma unroll
    for (int o = 0; o < CI; ++o) dst[o] = (_Float16)(acc[o] * scl);
  } else {
    _Float16* dst = hRM + (size_t)b * CI * NPIX + n;
#pragma unroll
    for (int o = 0; o < CI; ++o) dst[(size_t)o * NPIX] = (_Float16)acc[o];
  }
}

// ---------------- Phase 2: fused flash-attention + output projection ----------------
// Block = 8 waves = 128 query pixels. Per 32-key chunk the block stages g/h tiles into
// LDS once (async, double-buffered), then each wave:
//   2x v_wmma_f32_16x16x32_f16 for S^T tiles (m in VGPR index, n in lane),
//   online base-2 softmax (one xor-16 swizzle per reduction),
//   repack P^T into f16 B-layout, 2x WMMA for O^T += h x P^T.
// Tail: o -> C projection fused with y = gamma*o + x.
__global__ __launch_bounds__(256)
void attn_kernel(const _Float16* __restrict__ fT, const _Float16* __restrict__ gT,
                 const _Float16* __restrict__ hRM, const float* __restrict__ x,
                 const float* __restrict__ Wv, const float* __restrict__ bv,
                 const float* __restrict__ gamma, float* __restrict__ y) {
  // stage rows padded to 40 halves (80B): 16 lanes -> 16 disjoint 4-bank groups
  __shared__ __align__(16) _Float16 gs[2][32][40];
  __shared__ __align__(16) _Float16 hs[2][32][40];
  __shared__ float o_lds[CI][130];

  const int b    = blockIdx.y;
  const int nblk = blockIdx.x;                 // 128-pixel block
  const int wave = threadIdx.x >> 5;
  const int lane = threadIdx.x & 31;
  const int col  = lane & 15;
  const bool hi  = lane >= 16;

  const _Float16* fTb = fT  + (size_t)b * NPIX * CI;
  const _Float16* gTb = gT  + (size_t)b * NPIX * CI;
  const _Float16* hb  = hRM + (size_t)b * CI * NPIX;

  const int n0 = nblk * 128 + wave * 16;
  // B operand: f tile [32c x 16n]; lane = column n, lanes<16 K=0..15, lanes>=16 K=16..31
  const int koff = hi ? 16 : 0;
  v16h Bf = *(const v16h*)(fTb + (size_t)(n0 + col) * CI + koff);

  // staging role of this thread: t<128 -> g rows, t>=128 -> h rows; one 16B seg each
  const int trow  = (threadIdx.x & 127) >> 2;
  const int tseg  = threadIdx.x & 3;
  const bool is_h = threadIdx.x >= 128;

  const int aoff = hi ? 16 : 0;                // 16-bit A-layout half offset
  v8f O0 = {}, O1 = {};
  float run_max = -3.0e38f, run_den = 0.0f;
  const v8f zc = {};

  // prologue: stage chunk 0 into buffer 0
  {
    const _Float16* src = is_h ? (hb + (size_t)trow * NPIX + tseg * 8)
                               : (gTb + (size_t)trow * CI + tseg * 8);
    _Float16* dst = is_h ? &hs[0][trow][tseg * 8] : &gs[0][trow][tseg * 8];
    copy16(src, dst);
    copy_fence();
  }

  for (int m0 = 0; m0 < NPIX; m0 += 32) {
    const int buf = (m0 >> 5) & 1;
    // issue async staging of the next chunk into the other buffer
    if (m0 + 32 < NPIX) {
      const int mn = m0 + 32;
      const _Float16* src = is_h ? (hb + (size_t)trow * NPIX + mn + tseg * 8)
                                 : (gTb + (size_t)(mn + trow) * CI + tseg * 8);
      _Float16* dst = is_h ? &hs[buf ^ 1][trow][tseg * 8] : &gs[buf ^ 1][trow][tseg * 8];
      copy16(src, dst);
    }

    // A operands from LDS: g^T rows m (two 16-row tiles)
    v16h Ag0 = cat16(*(const v8h*)&gs[buf][col][aoff],
                     *(const v8h*)&gs[buf][col][aoff + 8]);
    v16h Ag1 = cat16(*(const v8h*)&gs[buf][col + 16][aoff],
                     *(const v8h*)&gs[buf][col + 16][aoff + 8]);
    // S^T[m,n]: D layout -> VGPR index = m (softmax axis), lane = n (base-2 logits)
    v8f T0 = __builtin_amdgcn_wmma_f32_16x16x32_f16(false, Ag0, false, Bf, (short)0, zc, false, false);
    v8f T1 = __builtin_amdgcn_wmma_f32_16x16x32_f16(false, Ag1, false, Bf, (short)0, zc, false, false);

    // online softmax over m for each column n
    float cm = T0[0];
#pragma unroll
    for (int r = 1; r < 8; ++r) cm = fmaxf(cm, T0[r]);
#pragma unroll
    for (int r = 0; r < 8; ++r) cm = fmaxf(cm, T1[r]);
    cm = fmaxf(cm, swap16f(cm));               // combine both m-halves
    float nm   = fmaxf(run_max, cm);
    float corr = fast_exp2(run_max - nm);
    float P0[8], P1[8];
    float cs = 0.0f;
#pragma unroll
    for (int r = 0; r < 8; ++r) { P0[r] = fast_exp2(T0[r] - nm); cs += P0[r]; }
#pragma unroll
    for (int r = 0; r < 8; ++r) { P1[r] = fast_exp2(T1[r] - nm); cs += P1[r]; }
    cs += swap16f(cs);
    run_den = run_den * corr + cs;
    run_max = nm;
#pragma unroll
    for (int r = 0; r < 8; ++r) { O0[r] *= corr; O1[r] *= corr; }

    // repack P^T (D layout) into f16 B-layout [32m x 16n] via xor-16 swaps
    float s0[8], s1[8];
#pragma unroll
    for (int r = 0; r < 8; ++r) { s0[r] = swap16f(P0[r]); s1[r] = swap16f(P1[r]); }
    v16h BP;
#pragma unroll
    for (int v = 0; v < 4; ++v) {
      BP[2 * v]         = (_Float16)(hi ? s1[2 * v]     : P0[2 * v]);
      BP[2 * v + 1]     = (_Float16)(hi ? s1[2 * v + 1] : P0[2 * v + 1]);
      BP[8 + 2 * v]     = (_Float16)(hi ? P1[2 * v]     : s0[2 * v]);
      BP[8 + 2 * v + 1] = (_Float16)(hi ? P1[2 * v + 1] : s0[2 * v + 1]);
    }
    // A operand: h rows c from LDS, O^T += h x P^T
    v16h Ah0 = cat16(*(const v8h*)&hs[buf][col][aoff],
                     *(const v8h*)&hs[buf][col][aoff + 8]);
    v16h Ah1 = cat16(*(const v8h*)&hs[buf][col + 16][aoff],
                     *(const v8h*)&hs[buf][col + 16][aoff + 8]);
    O0 = __builtin_amdgcn_wmma_f32_16x16x32_f16(false, Ah0, false, BP, (short)0, O0, false, false);
    O1 = __builtin_amdgcn_wmma_f32_16x16x32_f16(false, Ah1, false, BP, (short)0, O1, false, false);

    // own async issues done + all waves' staging visible before next iteration
    copy_fence();
  }

  // normalize and stage O^T[ci][n_local] in LDS
  const float inv = 1.0f / run_den;
  const int wcol = wave * 16 + col;
  const int chi  = hi ? 8 : 0;
#pragma unroll
  for (int r = 0; r < 8; ++r) {
    o_lds[r + chi][wcol]      = O0[r] * inv;
    o_lds[16 + r + chi][wcol] = O1[r] * inv;
  }
  __syncthreads();

  // output projection + residual: y = gamma*(Wv o + bv) + x   (coalesced per-co rows)
  const float gm = gamma[0];
  const int n_l = threadIdx.x & 127;
  const int co0 = (threadIdx.x >> 7) * 128;
  float ov[CI];
#pragma unroll
  for (int ci = 0; ci < CI; ++ci) ov[ci] = o_lds[ci][n_l];
  for (int co = co0; co < co0 + 128; ++co) {
    float acc = bv[co];
#pragma unroll
    for (int ci = 0; ci < CI; ++ci) acc = fmaf(Wv[co * CI + ci], ov[ci], acc);
    size_t idx = ((size_t)(b * CC + co)) * NPIX + (size_t)nblk * 128 + n_l;
    y[idx] = fmaf(gm, acc, x[idx]);
  }
}

extern "C" void kernel_launch(void* const* d_in, const int* in_sizes, int n_in,
                              void* d_out, int out_size, void* d_ws, size_t ws_size,
                              hipStream_t stream) {
  const float* x     = (const float*)d_in[0];
  const float* Wf    = (const float*)d_in[1];
  const float* bf    = (const float*)d_in[2];
  const float* Wg    = (const float*)d_in[3];
  const float* bg    = (const float*)d_in[4];
  const float* Wh    = (const float*)d_in[5];
  const float* bh    = (const float*)d_in[6];
  const float* Wv    = (const float*)d_in[7];
  const float* bv    = (const float*)d_in[8];
  const float* gamma = (const float*)d_in[9];
  float* y = (float*)d_out;

  char* ws = (char*)d_ws;
  _Float16* fT  = (_Float16*)(ws);                    // 1 MB
  _Float16* gT  = (_Float16*)(ws + (1u << 20));       // 1 MB
  _Float16* hRM = (_Float16*)(ws + (2u << 20));       // 1 MB

  proj_kernel<<<dim3(NPIX / 256, 3, BN), 256, 0, stream>>>(
      x, Wf, bf, Wg, bg, Wh, bh, fT, gT, hRM);
  attn_kernel<<<dim3(NPIX / 128, BN), 256, 0, stream>>>(
      fT, gT, hRM, x, Wv, bv, gamma, y);
}
